// NTXentLoss_67310727463046
// MI455X (gfx1250) — matrix operand order repaired and checked by reading, not compiled, not run on hardware
//
#include <hip/hip_runtime.h>
#include <hip/hip_bf16.h>
#include <math.h>

// NT-Xent loss, fused: normalize -> f16 WMMA GEMM (S = Zn * Zn^T) with fused
// exp/rowsum/positive epilogue -> final log/mean reduction.
// GEMM staging uses the CDNA5 Tensor Data Mover (tensor_load_to_lds) with
// double-buffered LDS tiles; falls back to manual staging if the TDM builtin
// is unavailable.
//
// Workspace layout (needs ~8.4 MB):
//   [0, 16KB)      float expsum[4096]
//   [16KB, 32KB)   float pos[4096]
//   [32KB, ~8.4MB) _Float16 zn[4096*1024]

typedef __attribute__((ext_vector_type(16))) _Float16 v16h;
typedef __attribute__((ext_vector_type(8)))  float    v8f;
typedef __attribute__((ext_vector_type(4)))  float    f4;
typedef __attribute__((ext_vector_type(4)))  _Float16 h4;
typedef __attribute__((ext_vector_type(4)))  unsigned int u32x4;
typedef __attribute__((ext_vector_type(8)))  int      i32x8;
typedef __attribute__((ext_vector_type(4)))  int      i32x4;

union Frag { v16h h; f4 f[2]; };

#define BATCH  2048
#define ROWS   4096   // 2*BATCH
#define DIM    1024
#define INVT   2.0f   // 1/TEMP

#define KCH    64     // K chunk staged in LDS
#define LDSP   72     // padded LDS row stride in halves (conflict-free b128)
#define NKC    (DIM / KCH)

#if defined(__gfx1250__) && __has_builtin(__builtin_amdgcn_tensor_load_to_lds)
#define USE_TDM 1
#else
#define USE_TDM 0
#endif

// ---------------------------------------------------------------- helpers
__device__ __forceinline__ void wait_tensorcnt0() {
#if defined(__gfx1250__)
#if __has_builtin(__builtin_amdgcn_s_wait_tensorcnt)
    __builtin_amdgcn_s_wait_tensorcnt((short)0);
#else
    asm volatile("s_wait_tensorcnt 0x0" ::: "memory");
#endif
#endif
}

#if USE_TDM
// Issue one TDM load of a 128-row x 64-half tile of zn into LDS.
// D# layout per CDNA5 ISA:
//  g0: count=1 | lds_addr | global_addr(57b) | type=2
//  g1: data_size=2B, pad_enable, pad_interval=32 DWORDs (code 4),
//      pad_amount=4 DWORDs (code 3) -> LDS row stride 72 halves,
//      tensor_dim0=1024, tensor_dim1=4096, tile_dim0=64, tile_dim1=128,
//      tensor_dim0_stride=1024
__device__ __forceinline__ void tdm_tile_load(const _Float16* gtile,
                                              unsigned int lds_byte) {
    unsigned long long ga = (unsigned long long)(uintptr_t)gtile;
    u32x4 g0;
    g0[0] = 1u;                                        // count=1, user mode
    g0[1] = lds_byte;                                  // LDS byte address
    g0[2] = (unsigned int)ga;                          // global addr lo
    g0[3] = (unsigned int)((ga >> 32) & 0x1FFFFFFu) | (2u << 30); // hi | type=2
    i32x8 g1;
    g1[0] = (1 << 16)          // data_size = 2 bytes
          | (1 << 20)          // pad_enable
          | (4 << 22)          // pad_interval: 32 DWORDs (128B) per row
          | (3 << 25);         // pad_amount: 4 DWORDs (16B)
    g1[1] = (int)((DIM & 0xFFFF) << 16);                       // tdim0 lo16
    g1[2] = (int)(((DIM >> 16) & 0xFFFF) | ((ROWS & 0xFFFF) << 16)); // tdim0 hi, tdim1 lo
    g1[3] = (int)(((ROWS >> 16) & 0xFFFF) | (KCH << 16));      // tdim1 hi, tile_dim0=64
    g1[4] = 128;               // tile_dim1 = 128 rows (tile_dim2 = 0)
    g1[5] = DIM;               // tensor_dim0_stride lo32 = 1024
    g1[6] = 0;                 // stride hi, tensor_dim1_stride (unused, 2D)
    g1[7] = 0;
    i32x4 z4 = {0, 0, 0, 0};               // groups 2/3 unused for 2D tiles
    i32x8 z8 = {0, 0, 0, 0, 0, 0, 0, 0};   // 6-arg toolchain variant
    __builtin_amdgcn_tensor_load_to_lds(g0, g1, z4, z4, z8, 0);
}
#endif

// ---------------------------------------------------------------- zero
__global__ void ntx_zero_kernel(float* p, int n) {
    int i = blockIdx.x * blockDim.x + threadIdx.x;
    if (i < n) p[i] = 0.0f;
}

// ---------------------------------------------------------------- normalize
__global__ void ntx_normalize_kernel(const float* __restrict__ z1,
                                     const float* __restrict__ z2,
                                     _Float16* __restrict__ zn) {
    const int row = blockIdx.x;
    const float* src = (row < BATCH) ? (z1 + (size_t)row * DIM)
                                     : (z2 + (size_t)(row - BATCH) * DIM);
    f4 v = ((const f4*)src)[threadIdx.x];
    float ss = v[0]*v[0] + v[1]*v[1] + v[2]*v[2] + v[3]*v[3];

    #pragma unroll
    for (int m = 1; m < 32; m <<= 1) ss += __shfl_xor(ss, m, 32);

    __shared__ float red[8];
    if ((threadIdx.x & 31) == 0) red[threadIdx.x >> 5] = ss;
    __syncthreads();
    float tot = 0.0f;
    #pragma unroll
    for (int i = 0; i < 8; ++i) tot += red[i];

    const float scale = 1.0f / fmaxf(sqrtf(tot), 1e-12f);
    h4 o;
    #pragma unroll
    for (int i = 0; i < 4; ++i) o[i] = (_Float16)(v[i] * scale);
    ((h4*)(zn + (size_t)row * DIM))[threadIdx.x] = o;
}

// ---------------------------------------------------------------- fused GEMM
// Grid: (32, 32) blocks of 256 threads (8 waves). Block (bx,by) computes the
// 128x128 tile S[by*128.., bx*128..] and folds it into per-row exp-sums and
// positive extraction. S never touches HBM; Zn (8MB f16) lives in L2.
__global__ __launch_bounds__(256)
void ntx_gemm_kernel(const _Float16* __restrict__ zn,
                     float* __restrict__ expsum,
                     float* __restrict__ pos) {
#if USE_TDM
    __shared__ _Float16 lAs[2][128][LDSP];
    __shared__ _Float16 lBs[2][128][LDSP];
#else
    __shared__ _Float16 lAs[1][128][LDSP];
    __shared__ _Float16 lBs[1][128][LDSP];
#endif

    const int t      = threadIdx.x;
    const int wave   = t >> 5;
    const int lane   = t & 31;
    const int hi     = lane >> 4;
    const int ln     = lane & 15;
    const int m_base = blockIdx.y * 128;
    const int n_base = blockIdx.x * 128;

    v8f acc[8] = {};

#if USE_TDM
    const bool issuer = (wave == 0);
    if (issuer) {
        tdm_tile_load(zn + (size_t)m_base * DIM,
                      (unsigned int)(uintptr_t)&lAs[0][0][0]);
        tdm_tile_load(zn + (size_t)n_base * DIM,
                      (unsigned int)(uintptr_t)&lBs[0][0][0]);
    }
#endif

    for (int kc = 0; kc < NKC; ++kc) {
#if USE_TDM
        const int cur = kc & 1;
        if (issuer) wait_tensorcnt0();      // this buffer's DMA complete
        __syncthreads();                    // publish; prev compute finished
        if (issuer && (kc + 1) < NKC) {     // prefetch next chunk -> other buf
            const int k0n = (kc + 1) * KCH;
            tdm_tile_load(zn + (size_t)m_base * DIM + k0n,
                          (unsigned int)(uintptr_t)&lAs[cur ^ 1][0][0]);
            tdm_tile_load(zn + (size_t)n_base * DIM + k0n,
                          (unsigned int)(uintptr_t)&lBs[cur ^ 1][0][0]);
        }
#else
        const int cur = 0;
        const int k0  = kc * KCH;
        f4 ra[4], rb[4];
        #pragma unroll
        for (int j = 0; j < 4; ++j) {
            int c    = t + 256 * j;
            int row  = c >> 3;
            int hoff = (c & 7) << 3;
            ra[j] = *(const f4*)(zn + (size_t)(m_base + row) * DIM + k0 + hoff);
            rb[j] = *(const f4*)(zn + (size_t)(n_base + row) * DIM + k0 + hoff);
        }
        __syncthreads();
        #pragma unroll
        for (int j = 0; j < 4; ++j) {
            int c    = t + 256 * j;
            int row  = c >> 3;
            int hoff = (c & 7) << 3;
            *(f4*)&lAs[0][row][hoff] = ra[j];
            *(f4*)&lBs[0][row][hoff] = rb[j];
        }
        __syncthreads();
#endif

        const _Float16 (*As)[LDSP] = lAs[cur];
        const _Float16 (*Bs)[LDSP] = lBs[cur];

        #pragma unroll
        for (int kk = 0; kk < KCH; kk += 32) {
            // A fragment (16x32 f16, ISA layout): lane half selects K groups
            Frag a;
            const int arow = wave * 16 + ln;
            const int koff = kk + hi * 8;
            a.f[0] = *(const f4*)&As[arow][koff];
            a.f[1] = *(const f4*)&As[arow][koff + 16];

            #pragma unroll
            for (int sub = 0; sub < 8; ++sub) {
                // B fragment (32x16, ISA layout): lane half selects K groups
                Frag b;
                const int brow = sub * 16 + ln;
                const int bk   = kk + hi * 8;
                b.f[0] = *(const f4*)&Bs[brow][bk];
                b.f[1] = *(const f4*)&Bs[brow][bk + 16];

                acc[sub] = __builtin_amdgcn_wmma_f32_16x16x32_f16(
                    false, a.h, false, b.h, (short)0, acc[sub], false, false);
            }
        }

#if !USE_TDM
        __syncthreads();   // single buffer: protect LDS before next staging
#endif
    }

    // ---- fused epilogue: exp row-sums (excluding diagonal) + positives.
    float es[8], ps[8];
    #pragma unroll
    for (int r = 0; r < 8; ++r) { es[r] = 0.0f; ps[r] = 0.0f; }

    #pragma unroll
    for (int sub = 0; sub < 8; ++sub) {
        const int col = n_base + sub * 16 + ln;
        #pragma unroll
        for (int r = 0; r < 8; ++r) {
            const int row = m_base + wave * 16 + hi * 8 + r;
            const float s = acc[sub][r];
            if (col != row)           es[r] += __expf(s * INVT);
            if (col == (row ^ BATCH)) ps[r] += s;
        }
    }

    #pragma unroll
    for (int m = 1; m < 16; m <<= 1) {
        #pragma unroll
        for (int r = 0; r < 8; ++r) {
            es[r] += __shfl_xor(es[r], m, 32);
            ps[r] += __shfl_xor(ps[r], m, 32);
        }
    }

    if (ln == 0) {
        #pragma unroll
        for (int r = 0; r < 8; ++r) {
            const int row = m_base + wave * 16 + hi * 8 + r;
            atomicAdd(&expsum[row], es[r]);
            atomicAdd(&pos[row],    ps[r]);
        }
    }
}

// ---------------------------------------------------------------- finalize
__global__ void ntx_finalize_kernel(const float* __restrict__ expsum,
                                    const float* __restrict__ pos,
                                    float* __restrict__ out) {
    float local = 0.0f;
    for (int i = threadIdx.x; i < ROWS; i += 256)
        local += pos[i] * INVT - logf(expsum[i]);

    #pragma unroll
    for (int m = 1; m < 32; m <<= 1) local += __shfl_xor(local, m, 32);

    __shared__ float red[8];
    if ((threadIdx.x & 31) == 0) red[threadIdx.x >> 5] = local;
    __syncthreads();
    if (threadIdx.x == 0) {
        float tot = 0.0f;
        #pragma unroll
        for (int i = 0; i < 8; ++i) tot += red[i];
        out[0] = -tot / (float)ROWS;
    }
}

// ---------------------------------------------------------------- launch
extern "C" void kernel_launch(void* const* d_in, const int* in_sizes, int n_in,
                              void* d_out, int out_size, void* d_ws, size_t ws_size,
                              hipStream_t stream) {
    const float* z1 = (const float*)d_in[0];
    const float* z2 = (const float*)d_in[1];

    float*     expsum = (float*)d_ws;
    float*     pos    = expsum + ROWS;
    _Float16*  zn     = (_Float16*)((char*)d_ws + 32768);
    float*     out    = (float*)d_out;

    ntx_zero_kernel<<<8, 1024, 0, stream>>>(expsum, 2 * ROWS);
    ntx_normalize_kernel<<<ROWS, 256, 0, stream>>>(z1, z2, zn);
    dim3 grid(ROWS / 128, ROWS / 128);
    ntx_gemm_kernel<<<grid, 256, 0, stream>>>(zn, expsum, pos);
    ntx_finalize_kernel<<<1, 256, 0, stream>>>(expsum, pos, out);
}
